// ChannelAttention_42614665511016
// MI455X (gfx1250) — matrix operand order repaired and checked by reading, not compiled, not run on hardware
//
#include <hip/hip_runtime.h>

#define NCH 256
#define HWS 16384
#define NB  8
#define NH  8
#define DH  32

typedef __attribute__((ext_vector_type(16))) __bf16 v16bf;
typedef __attribute__((ext_vector_type(8)))  float  v8f;

union Frag {
  uint4 u4[2];
  v16bf bf;
};

__device__ __forceinline__ unsigned packbf2(float a, float b) {
  union { __bf16 h[2]; unsigned u; } c;
  c.h[0] = (__bf16)a; c.h[1] = (__bf16)b;
  return c.u;
}

__device__ __forceinline__ void gatomicAdd(float* p, float v) {
  __hip_atomic_fetch_add(p, v, __ATOMIC_RELAXED, __HIP_MEMORY_SCOPE_AGENT);
}

__global__ __launch_bounds__(256) void k_zero(float* __restrict__ p, int n) {
  int i = blockIdx.x * 256 + threadIdx.x;
  if (i < n) p[i] = 0.f;
}

// ---------------------------------------------------------------------------
// K1: S_b = x_b * x_b^T  (256x256 per batch), bf16 WMMA, fp32 accumulate.
// grid = (kc=4, rowgroup=4, batch=8); block = 256 (8 waves).
// Double-buffered LDS: one barrier per k-step; next tile's global loads are
// issued before the WMMA stream so LOADcnt drains behind the matrix ops.
// ---------------------------------------------------------------------------
__global__ __launch_bounds__(256) void k_gram(const float* __restrict__ x,
                                              float* __restrict__ S) {
  const int b  = blockIdx.z;
  const int rg = blockIdx.y;
  const int kc = blockIdx.x;
  const int t = threadIdx.x, lane = t & 31, wave = t >> 5;
  const int hi = lane >> 4;

  __shared__ __align__(16) unsigned tileU[2][NCH][16];  // 2 x [chan][32 bf16], 32KB

  const float* xb = x + (size_t)b * NCH * HWS;
  const float* xrow = xb + (size_t)t * HWS;
  float* Sb = S + (size_t)b * NCH * NCH;

  const int rowtile = rg * 64 + (wave >> 1) * 16;  // this wave's A row tile
  const int ctbase  = (wave & 1) * 8;              // 8 col tiles per wave

  const v8f vzero = {0.f,0.f,0.f,0.f,0.f,0.f,0.f,0.f};
  v8f acc[8];
  #pragma unroll
  for (int i = 0; i < 8; ++i) acc[i] = vzero;

  const int k0beg = kc * 4096;
  const int k0end = k0beg + 4096;

  // prologue: stage first tile into buffer 0
  {
    const float4* src = (const float4*)(xrow + k0beg);
    #pragma unroll
    for (int i = 0; i < 8; ++i) {
      float4 v = src[i];
      tileU[0][t][i*2+0] = packbf2(v.x, v.y);
      tileU[0][t][i*2+1] = packbf2(v.z, v.w);
    }
  }

  int cur = 0;
  for (int k0 = k0beg; k0 < k0end; k0 += 32) {
    __syncthreads();  // tile[cur] staged; everyone done reading tile[cur^1]
    const bool more = (k0 + 32) < k0end;

    // issue next tile's global loads now (no dependency on LDS compute below)
    float4 f[8];
    if (more) {
      const float4* src = (const float4*)(xrow + k0 + 32);
      #pragma unroll
      for (int i = 0; i < 8; ++i) f[i] = src[i];
    }

    // A fragment: 16 rows x 32 K (ISA 7.12.2 16-bit A layout)
    Frag fa;
    { const uint4* rp = (const uint4*)tileU[cur][rowtile + (lane & 15)];
      fa.u4[0] = rp[hi]; fa.u4[1] = rp[2 + hi]; }
    #pragma unroll
    for (int c = 0; c < 8; ++c) {
      // B fragment: 32 K x 16 cols (lane = N, K in same table order as A)
      Frag fb;
      const uint4* rp = (const uint4*)tileU[cur][(ctbase + c) * 16 + (lane & 15)];
      fb.u4[0] = rp[hi]; fb.u4[1] = rp[2 + hi];
      acc[c] = __builtin_amdgcn_wmma_f32_16x16x32_bf16(
          false, fa.bf, false, fb.bf, (short)0, acc[c], false, false);
    }

    // convert & stage next tile into the other buffer (after the WMMA stream)
    if (more) {
      #pragma unroll
      for (int i = 0; i < 8; ++i) {
        tileU[cur ^ 1][t][i*2+0] = packbf2(f[i].x, f[i].y);
        tileU[cur ^ 1][t][i*2+1] = packbf2(f[i].z, f[i].w);
      }
    }
    cur ^= 1;
  }

  #pragma unroll
  for (int c = 0; c < 8; ++c) {
    const int col = (ctbase + c) * 16 + (lane & 15);
    #pragma unroll
    for (int r = 0; r < 8; ++r) {
      const int row = rowtile + r + hi * 8;  // C/D layout: VGPR r -> M=r / r+8
      gatomicAdd(&Sb[row * NCH + col], acc[c][r]);
    }
  }
}

// ---------------------------------------------------------------------------
// K2: per (b,h): G = Wq_h S Wk_h^T, norms from diagonals, softmax*temp,
//     P_h = attn * Wv_h  -> P_b[h*32+i, :].  Pure fp32.
// grid = (h=8, b=8); block = 256 (thread t = column index).
// ---------------------------------------------------------------------------
__global__ __launch_bounds__(256) void k_attn(const float* __restrict__ S,
                                              const float* __restrict__ wqkv,
                                              const float* __restrict__ temp,
                                              float* __restrict__ P) {
  const int h = blockIdx.x, b = blockIdx.y, t = threadIdx.x;

  __shared__ float WkL[DH * NCH];    // 32KB
  __shared__ float attnL[DH * DH];   // 4KB (G, then probs)
  __shared__ float ssqq[DH], ssqk[DH];

  const float* Sb  = S + (size_t)b * NCH * NCH;
  const float* Wqg = wqkv + (size_t)(h * DH) * NCH;
  const float* Wkg = wqkv + (size_t)(NCH + h * DH) * NCH;
  const float* Wvg = wqkv + (size_t)(2 * NCH + h * DH) * NCH;

  for (int i = t; i < DH * NCH; i += 256) WkL[i] = Wkg[i];
  for (int i = t; i < DH * DH; i += 256) attnL[i] = 0.f;
  if (t < DH) { ssqq[t] = 0.f; ssqk[t] = 0.f; }
  __syncthreads();

  // Aq[i,t] = (Wq_h S)[i,t], Ak[i,t] = (Wk_h S)[i,t]
  float aq[DH], ak[DH];
  #pragma unroll
  for (int i = 0; i < DH; ++i) { aq[i] = 0.f; ak[i] = 0.f; }
  for (int m = 0; m < NCH; ++m) {
    const float s = Sb[m * NCH + t];
    #pragma unroll
    for (int i = 0; i < DH; ++i) {
      aq[i] += Wqg[i * NCH + m] * s;   // uniform -> scalar loads
      ak[i] += WkL[i * NCH + m] * s;   // uniform LDS broadcast
    }
  }
  // ssq_q[i] = diag(Wq S Wq^T), ssq_k likewise (S symmetric)
  #pragma unroll
  for (int i = 0; i < DH; ++i) {
    atomicAdd(&ssqq[i], aq[i] * Wqg[i * NCH + t]);
    atomicAdd(&ssqk[i], ak[i] * WkL[i * NCH + t]);
  }
  // G[i,d] = sum_t Aq[i,t] * Wk_h[d,t]
  for (int d = 0; d < DH; ++d) {
    const float wk = WkL[d * NCH + t];
    #pragma unroll
    for (int i = 0; i < DH; ++i) atomicAdd(&attnL[i * DH + d], aq[i] * wk);
  }
  __syncthreads();

  if (t < DH) {  // softmax row t (F.normalize eps semantics)
    const float tv = temp[h];
    const float nq = fmaxf(sqrtf(fmaxf(ssqq[t], 0.f)), 1e-12f);
    float row[DH];
    float mx = -3.4e38f;
    for (int d = 0; d < DH; ++d) {
      const float nk = fmaxf(sqrtf(fmaxf(ssqk[d], 0.f)), 1e-12f);
      const float l = attnL[t * DH + d] * tv / (nq * nk);
      row[d] = l; mx = fmaxf(mx, l);
    }
    float ssum = 0.f;
    for (int d = 0; d < DH; ++d) { const float e = __expf(row[d] - mx); row[d] = e; ssum += e; }
    const float inv = 1.f / ssum;
    for (int d = 0; d < DH; ++d) attnL[t * DH + d] = row[d] * inv;
  }
  __syncthreads();

  // P_h[i, t] = sum_d attn[i,d] * Wv_h[d, t]
  float ap[DH];
  #pragma unroll
  for (int i = 0; i < DH; ++i) ap[i] = 0.f;
  for (int d = 0; d < DH; ++d) {
    const float wv = Wvg[d * NCH + t];
    #pragma unroll
    for (int i = 0; i < DH; ++i) ap[i] += attnL[i * DH + d] * wv;
  }
  float* Pb = P + (size_t)b * NCH * NCH;
  #pragma unroll
  for (int i = 0; i < DH; ++i) Pb[(h * DH + i) * NCH + t] = ap[i];
}

// ---------------------------------------------------------------------------
// K2b: M_b = w_proj @ P_b, stored packed bf16 (row-major, 2 per dword).
// grid = (coltile=16, b=8); block = 256 (thread t = output row).
// ---------------------------------------------------------------------------
__global__ __launch_bounds__(256) void k_combine(const float* __restrict__ wproj,
                                                 const float* __restrict__ P,
                                                 unsigned* __restrict__ Mbf) {
  const int b = blockIdx.y;
  const int c0 = blockIdx.x * 16;
  const int t = threadIdx.x;
  __shared__ float Pl[NCH][16];
  const float* Pb = P + (size_t)b * NCH * NCH;
  #pragma unroll
  for (int c = 0; c < 16; ++c) Pl[t][c] = Pb[t * NCH + c0 + c];
  __syncthreads();
  float accv[16];
  #pragma unroll
  for (int c = 0; c < 16; ++c) accv[c] = 0.f;
  for (int m = 0; m < NCH; ++m) {
    const float wp = wproj[t * NCH + m];
    #pragma unroll
    for (int c = 0; c < 16; ++c) accv[c] += wp * Pl[m][c];
  }
  unsigned* Mb = Mbf + (size_t)b * (NCH * NCH / 2) + t * (NCH / 2) + (c0 >> 1);
  #pragma unroll
  for (int c = 0; c < 8; ++c) Mb[c] = packbf2(accv[2*c], accv[2*c+1]);
}

// ---------------------------------------------------------------------------
// K3: y_b = M_b @ x_b, bf16 WMMA. grid = (HW/128, b=8); block = 256 (8 waves).
// x tile transposed into double-buffered LDS [spatial][chan]; A fragments
// read straight from packed-bf16 M in global (L2/WGP$ resident).
// ---------------------------------------------------------------------------
__global__ __launch_bounds__(256) void k_apply(const float* __restrict__ x,
                                               const unsigned* __restrict__ Mbf,
                                               float* __restrict__ y) {
  const int b = blockIdx.y;
  const int s0 = blockIdx.x * 128;
  const int t = threadIdx.x, lane = t & 31, wave = t >> 5;
  const int hi = lane >> 4;

  __shared__ __align__(16) __bf16 ldsX[2][128 * 32];  // 2 x [spatial][chan], 16KB

  const float* xb = x + (size_t)b * NCH * HWS;
  const unsigned* Mb = Mbf + (size_t)b * (NCH * NCH / 2);

  const v8f vzero = {0.f,0.f,0.f,0.f,0.f,0.f,0.f,0.f};
  v8f acc[16];
  #pragma unroll
  for (int i = 0; i < 16; ++i) acc[i] = vzero;

  const int ch  = t >> 3;        // local channel 0..31
  const int seg = (t & 7) * 16;  // spatial segment start

  // prologue: stage k0 = 0 into buffer 0
  {
    const float4* xp = (const float4*)(xb + (size_t)ch * HWS + s0 + seg);
    #pragma unroll
    for (int i = 0; i < 4; ++i) {
      float4 v = xp[i];
      ldsX[0][(seg + i*4 + 0) * 32 + ch] = (__bf16)v.x;
      ldsX[0][(seg + i*4 + 1) * 32 + ch] = (__bf16)v.y;
      ldsX[0][(seg + i*4 + 2) * 32 + ch] = (__bf16)v.z;
      ldsX[0][(seg + i*4 + 3) * 32 + ch] = (__bf16)v.w;
    }
  }

  int cur = 0;
  for (int k0 = 0; k0 < NCH; k0 += 32) {
    __syncthreads();  // buffer cur staged; prior reads of cur^1 done
    const bool more = (k0 + 32) < NCH;

    // issue next x tile's global loads before the WMMA stream
    float vals[16];
    if (more) {
      const float4* xp = (const float4*)(xb + (size_t)(k0 + 32 + ch) * HWS + s0 + seg);
      #pragma unroll
      for (int i = 0; i < 4; ++i) {
        float4 v = xp[i];
        vals[i*4+0] = v.x; vals[i*4+1] = v.y; vals[i*4+2] = v.z; vals[i*4+3] = v.w;
      }
    }

    // B fragment: K = channels (contiguous in LDS), N = spatial col
    Frag fb;
    { const uint4* bp = (const uint4*)(ldsX[cur] + (wave * 16 + (lane & 15)) * 32 + hi * 8);
      fb.u4[0] = bp[0]; fb.u4[1] = bp[2]; }

    #pragma unroll
    for (int rt = 0; rt < 16; ++rt) {
      Frag fa;  // A fragment from packed bf16 M: row rt*16+(lane&15), K chans
      const uint4* mp = (const uint4*)(Mb + (rt * 16 + (lane & 15)) * (NCH / 2) + (k0 >> 1));
      fa.u4[0] = mp[hi]; fa.u4[1] = mp[2 + hi];
      acc[rt] = __builtin_amdgcn_wmma_f32_16x16x32_bf16(
          false, fa.bf, false, fb.bf, (short)0, acc[rt], false, false);
    }

    // transpose-store next tile into the other buffer after the WMMAs
    if (more) {
      #pragma unroll
      for (int j = 0; j < 16; ++j) ldsX[cur ^ 1][(seg + j) * 32 + ch] = (__bf16)vals[j];
    }
    cur ^= 1;
  }

  float* yb = y + (size_t)b * NCH * HWS;
  const int col = s0 + wave * 16 + (lane & 15);
  #pragma unroll
  for (int rt = 0; rt < 16; ++rt) {
    #pragma unroll
    for (int r = 0; r < 8; ++r) {
      yb[(size_t)(rt * 16 + r + hi * 8) * HWS + col] = acc[rt][r];
    }
  }
}

extern "C" void kernel_launch(void* const* d_in, const int* in_sizes, int n_in,
                              void* d_out, int out_size, void* d_ws, size_t ws_size,
                              hipStream_t stream) {
  const float* x     = (const float*)d_in[0];
  const float* wqkv  = (const float*)d_in[1];
  const float* wproj = (const float*)d_in[2];
  const float* temp  = (const float*)d_in[3];
  float* y = (float*)d_out;

  // workspace: S (2MB) | P (2MB) | Mbf packed bf16 (1MB)
  float* S = (float*)d_ws;
  float* P = S + (size_t)NB * NCH * NCH;
  unsigned* Mbf = (unsigned*)(P + (size_t)NB * NCH * NCH);

  k_zero<<<dim3((NB * NCH * NCH) / 256), 256, 0, stream>>>(S, NB * NCH * NCH);
  k_gram<<<dim3(4, 4, NB), 256, 0, stream>>>(x, S);
  k_attn<<<dim3(NH, NB), 256, 0, stream>>>(S, wqkv, temp, P);
  k_combine<<<dim3(16, NB), 256, 0, stream>>>(wproj, P, Mbf);
  k_apply<<<dim3(HWS / 128, NB), 256, 0, stream>>>(x, Mbf, y);
}
